// CoSasrec_54176717471937
// MI455X (gfx1250) — matrix operand order repaired and checked by reading, not compiled, not run on hardware
//
#include <hip/hip_runtime.h>

// ---------------------------------------------------------------------------
// SASRec forward for MI455X (gfx1250, wave32).  All GEMMs via
// v_wmma_f32_16x16x32_bf16 (bf16 operands, fp32 accumulate).  Residual
// stream, layernorm stats and softmax are fp32.  All WMMA operands are
// loaded with contiguous b128 vector loads: weights are stored transposed
// (N x K) and V is produced transposed (H x L per batch).
// ---------------------------------------------------------------------------

typedef __bf16 bf16;
typedef __attribute__((ext_vector_type(8)))  bf16  v8bf;
typedef __attribute__((ext_vector_type(16))) bf16  v16bf;
typedef __attribute__((ext_vector_type(8)))  float v8f;

constexpr int kB  = 256;
constexpr int kL  = 256;
constexpr int kH  = 128;
constexpr int kNL = 2;
constexpr int kBL = kB * kL;
constexpr float kScale = 0.08838834764831845f;   // 1/sqrt(128)
constexpr float kSqrtH = 11.313708498984761f;    // sqrt(128)

static __device__ __forceinline__ v16bf cat8(v8bf lo, v8bf hi) {
  return __builtin_shufflevector(lo, hi, 0,1,2,3,4,5,6,7,8,9,10,11,12,13,14,15);
}

// A fragment: 16x32 bf16 tile of row-major A (leading dim lda).
// ISA layout: lanes 0-15 -> M=lane&15, K = kb..kb+7 and kb+16..kb+23, kb=k0;
//             lanes 16-31 -> same M, kb = k0+8.
static __device__ __forceinline__ v16bf load_a_frag(const bf16* A, int lda,
                                                    int m0, int k0, int lane) {
  int m  = m0 + (lane & 15);
  int kb = k0 + ((lane >> 4) << 3);
  const bf16* p = A + (size_t)m * lda + kb;
  v8bf lo = *(const v8bf*)p;          // K = kb..kb+7
  v8bf hi = *(const v8bf*)(p + 16);   // K = kb+16..kb+23
  return cat8(lo, hi);
}

// B fragment where the B matrix is available transposed, i.e. B[k][n] = T[n][k]
// with T row-major (N x K, leading dim ldk).  Per-lane loads are contiguous:
// element e -> T[n0 + (lane&15)][k0 + (lane>=16?8:0) + (e<8 ? e : e+8)]
static __device__ __forceinline__ v16bf load_b_frag_rows(const bf16* T, int ldk,
                                                         int k0, int n0, int lane) {
  int n  = n0 + (lane & 15);
  int ks = k0 + ((lane >> 4) << 3);
  const bf16* p = T + (size_t)n * ldk + ks;
  v8bf lo = *(const v8bf*)p;
  v8bf hi = *(const v8bf*)(p + 16);
  return cat8(lo, hi);
}

static __device__ __forceinline__ v8f wmma_bf16(v16bf a, v16bf b, v8f c) {
  return __builtin_amdgcn_wmma_f32_16x16x32_bf16(false, a, false, b,
                                                 (short)0, c, false, false);
}

// ---------------------------------------------------------------------------
// fp32 -> bf16 convert + transpose: Wt[layer][n][k] = W[layer][k][n]
// ---------------------------------------------------------------------------
__global__ void cvt_transpose_kernel(const float* __restrict__ W,
                                     bf16* __restrict__ Wt, int n_total) {
  int i = blockIdx.x * blockDim.x + threadIdx.x;
  if (i >= n_total) return;
  int layer = i / (kH * kH);
  int rem   = i - layer * (kH * kH);
  int k = rem >> 7;          // row of W
  int n = rem & (kH - 1);    // col of W
  Wt[(size_t)layer * kH * kH + (size_t)n * kH + k] = (bf16)W[i];
}

// ---------------------------------------------------------------------------
// Embedding gather + time-interval "rotary" + positional table -> x (fp32)
// grid = B*L blocks, block = H threads
// ---------------------------------------------------------------------------
__global__ void embed_rope_kernel(const int* __restrict__ positives,
                                  const float* __restrict__ times,
                                  const float* __restrict__ item_emb,
                                  const float* __restrict__ pos_table,
                                  float* __restrict__ X) {
  int bl = blockIdx.x;           // b*L + l
  int d  = threadIdx.x;          // hidden index 0..127
  int b  = bl >> 8;
  int l  = bl & (kL - 1);
  int idx = positives[bl];
  const float* e = item_emb + (size_t)idx * kH;
  float xe = e[d & ~1];
  float xo = e[d | 1];
  float xd = (d & 1) ? xo : xe;
  float rh = (d & 1) ? xe : -xo;  // rotate-half: rh[2i]=-x[2i+1], rh[2i+1]=x[2i]

  // ti2[b][j] = 0.5*(timediff[2j] + timediff[2j+1]); timediff[0]=0
  const float* t = times + (size_t)b * kL;
  int j = d >> 1;                 // 0..63
  float s0 = (j == 0) ? 0.0f : (t[2 * j] - t[2 * j - 1]);
  float sinv = 0.5f * (s0 + t[2 * j + 1] - t[2 * j]);
  int jc = 64 + j;                // 64..127 (always > 0)
  float cosv = 0.5f * ((t[2 * jc] - t[2 * jc - 1]) + (t[2 * jc + 1] - t[2 * jc]));

  X[(size_t)bl * kH + d] =
      pos_table[(size_t)l * kH + d] + xd * cosv + rh * sinv + xd * kSqrtH;
}

// ---------------------------------------------------------------------------
// Row-wise layernorm over H=128; OUT = bf16 (intermediate) or float (final)
// grid = B*L blocks, block = 128 threads
// ---------------------------------------------------------------------------
template <typename OUT>
__global__ void ln_kernel(const float* __restrict__ X,
                          const float* __restrict__ g,
                          const float* __restrict__ bb,
                          OUT* __restrict__ Y) {
  int row = blockIdx.x;
  int d   = threadIdx.x;
  float v = X[(size_t)row * kH + d];
  __shared__ float sm[kH];
  sm[d] = v;
  __syncthreads();
#pragma unroll
  for (int s = kH / 2; s > 0; s >>= 1) {
    if (d < s) sm[d] += sm[d + s];
    __syncthreads();
  }
  float mean = sm[0] * (1.0f / kH);
  __syncthreads();
  float c = v - mean;
  sm[d] = c * c;
  __syncthreads();
#pragma unroll
  for (int s = kH / 2; s > 0; s >>= 1) {
    if (d < s) sm[d] += sm[d + s];
    __syncthreads();
  }
  float var = sm[0] * (1.0f / kH);
  float r = rsqrtf(var + 1e-5f);
  Y[(size_t)row * kH + d] = (OUT)(c * r * g[d] + bb[d]);
}

// ---------------------------------------------------------------------------
// Projection GEMM: C = A[M=BL, K=H] @ Wt^T + bias, bf16 out.
//   mode 0: row-major out (B*L, H)
//   mode 1: row-major out with ReLU
//   mode 2: per-batch transposed out Ct[b][n][l]  (used for V -> V^T)
// Each wave computes a 16x32 output strip (one A frag, two WMMAs per k-step).
// block = 128 (4 waves); grid = (BL/64, H/32)
// ---------------------------------------------------------------------------
__global__ void proj_gemm_kernel(const bf16* __restrict__ A,
                                 const bf16* __restrict__ Wt,
                                 const float* __restrict__ bias,
                                 bf16* __restrict__ C, int mode) {
  int lane = threadIdx.x & 31;
  int wave = threadIdx.x >> 5;
  int m0 = (blockIdx.x * 4 + wave) * 16;
  int n0 = blockIdx.y * 32;
  v8f acc0 = {}, acc1 = {};
#pragma unroll
  for (int k0 = 0; k0 < kH; k0 += 32) {
    v16bf a  = load_a_frag(A, kH, m0, k0, lane);
    v16bf b0 = load_b_frag_rows(Wt, kH, k0, n0, lane);
    v16bf b1 = load_b_frag_rows(Wt, kH, k0, n0 + 16, lane);
    acc0 = wmma_bf16(a, b0, acc0);
    acc1 = wmma_bf16(a, b1, acc1);
  }
  int mr = m0 + ((lane >> 4) << 3);
#pragma unroll
  for (int half = 0; half < 2; ++half) {
    v8f acc = half ? acc1 : acc0;
    int n = n0 + half * 16 + (lane & 15);
    float bv = bias[n];
#pragma unroll
    for (int r = 0; r < 8; ++r) {
      float v = acc[r] + bv;
      if (mode == 1) v = fmaxf(v, 0.0f);
      int m = mr + r;
      if (mode == 2) {
        int b = m >> 8, l = m & (kL - 1);
        C[(size_t)b * kH * kL + (size_t)n * kL + l] = (bf16)v;
      } else {
        C[(size_t)m * kH + n] = (bf16)v;
      }
    }
  }
}

// ---------------------------------------------------------------------------
// Residual GEMM: X[M=BL, H] += A @ Wt^T + bias   (fp32 accumulate)
// ---------------------------------------------------------------------------
__global__ void resid_gemm_kernel(const bf16* __restrict__ A,
                                  const bf16* __restrict__ Wt,
                                  const float* __restrict__ bias,
                                  float* __restrict__ X) {
  int lane = threadIdx.x & 31;
  int wave = threadIdx.x >> 5;
  int m0 = (blockIdx.x * 4 + wave) * 16;
  int n0 = blockIdx.y * 32;
  v8f acc0 = {}, acc1 = {};
#pragma unroll
  for (int k0 = 0; k0 < kH; k0 += 32) {
    v16bf a  = load_a_frag(A, kH, m0, k0, lane);
    v16bf b0 = load_b_frag_rows(Wt, kH, k0, n0, lane);
    v16bf b1 = load_b_frag_rows(Wt, kH, k0, n0 + 16, lane);
    acc0 = wmma_bf16(a, b0, acc0);
    acc1 = wmma_bf16(a, b1, acc1);
  }
  int mr = m0 + ((lane >> 4) << 3);
#pragma unroll
  for (int half = 0; half < 2; ++half) {
    v8f acc = half ? acc1 : acc0;
    int n = n0 + half * 16 + (lane & 15);
    float bv = bias[n];
#pragma unroll
    for (int r = 0; r < 8; ++r) {
      size_t idx = (size_t)(mr + r) * kH + n;
      X[idx] = X[idx] + acc[r] + bv;
    }
  }
}

// ---------------------------------------------------------------------------
// Scores: S_b[L,L] = Q_b[L,H] @ K_b[L,H]^T * scale  (per batch)
// grid = (B, L/16 qtiles, 4); wave -> kt = z*4+wave; skip fully-future tiles.
// Both operands load contiguously (rows of K give the transpose for free).
// ---------------------------------------------------------------------------
__global__ void scores_kernel(const bf16* __restrict__ Q,
                              const bf16* __restrict__ Km,
                              float* __restrict__ S) {
  int b  = blockIdx.x;
  int qt = blockIdx.y;
  int kt = blockIdx.z * 4 + (threadIdx.x >> 5);
  if (kt > qt) return;                        // entirely masked (causal)
  int lane = threadIdx.x & 31;
  const bf16* Qb = Q  + (size_t)b * kL * kH;
  const bf16* Kb = Km + (size_t)b * kL * kH;
  v8f acc = {};
#pragma unroll
  for (int k0 = 0; k0 < kH; k0 += 32) {
    v16bf a = load_a_frag(Qb, kH, qt * 16, k0, lane);
    v16bf bfrag = load_b_frag_rows(Kb, kH, k0, kt * 16, lane);
    acc = wmma_bf16(a, bfrag, acc);
  }
  float* Sb = S + (size_t)b * kL * kL;
  int n  = kt * 16 + (lane & 15);
  int mr = qt * 16 + ((lane >> 4) << 3);
#pragma unroll
  for (int r = 0; r < 8; ++r)
    Sb[(size_t)(mr + r) * kL + n] = acc[r] * kScale;
}

// ---------------------------------------------------------------------------
// Masked softmax over each score row (fp32 in, bf16 probs out).
// Mask rule (matches reference): disallowed = (k != q) && (pad[k] || k > q).
// grid = B*L, block = L threads
// ---------------------------------------------------------------------------
__global__ void softmax_kernel(const float* __restrict__ S,
                               const int* __restrict__ mask,
                               bf16* __restrict__ P) {
  int row = blockIdx.x;             // b*L + q
  int b = row >> 8;
  int q = row & (kL - 1);
  int k = threadIdx.x;
  bool pad = (mask[(size_t)b * kL + k] == 0);
  bool dis = (k != q) && (pad || (k > q));
  float raw = S[(size_t)row * kL + k];
  float v = dis ? -__builtin_inff() : raw;

  __shared__ float sm[kL];
  sm[k] = v;
  __syncthreads();
#pragma unroll
  for (int s = kL / 2; s > 0; s >>= 1) {
    if (k < s) sm[k] = fmaxf(sm[k], sm[k + s]);
    __syncthreads();
  }
  float mx = sm[0];
  __syncthreads();
  float e = dis ? 0.0f : expf(v - mx);
  sm[k] = e;
  __syncthreads();
#pragma unroll
  for (int s = kL / 2; s > 0; s >>= 1) {
    if (k < s) sm[k] += sm[k + s];
    __syncthreads();
  }
  float rsum = 1.0f / sm[0];
  P[(size_t)row * kL + k] = (bf16)(e * rsum);
}

// ---------------------------------------------------------------------------
// Attention output: O_b[L,H] = P_b[L,L] @ V_b   with V stored transposed
// (Vt: per-batch H x L), so B-fragments load contiguously.
// grid = (B, L/16 qtiles); 4 waves, each computes two H-tiles (wave, wave+4).
// K-loop truncated at the causal boundary (probs beyond q are zero).
// ---------------------------------------------------------------------------
__global__ void attnv_kernel(const bf16* __restrict__ P,
                             const bf16* __restrict__ Vt,
                             bf16* __restrict__ O) {
  int b  = blockIdx.x;
  int qt = blockIdx.y;
  int wave = threadIdx.x >> 5;
  int lane = threadIdx.x & 31;
  int na = wave * 16;             // H-tile 1
  int nb = (wave + 4) * 16;       // H-tile 2
  const bf16* Pb  = P  + (size_t)b * kL * kL;
  const bf16* Vtb = Vt + (size_t)b * kH * kL;
  int kmax = (qt + 1) * 16;       // keys beyond this have prob 0
  v8f acc0 = {}, acc1 = {};
  for (int k0 = 0; k0 < kmax; k0 += 32) {
    v16bf a  = load_a_frag(Pb, kL, qt * 16, k0, lane);
    v16bf b0 = load_b_frag_rows(Vtb, kL, k0, na, lane);
    v16bf b1 = load_b_frag_rows(Vtb, kL, k0, nb, lane);
    acc0 = wmma_bf16(a, b0, acc0);
    acc1 = wmma_bf16(a, b1, acc1);
  }
  int mr = qt * 16 + ((lane >> 4) << 3);
  bf16* Ob = O + (size_t)b * kL * kH;
#pragma unroll
  for (int half = 0; half < 2; ++half) {
    v8f acc = half ? acc1 : acc0;
    int n = (half ? nb : na) + (lane & 15);
#pragma unroll
    for (int r = 0; r < 8; ++r)
      Ob[(size_t)(mr + r) * kH + n] = (bf16)acc[r];
  }
}

// ---------------------------------------------------------------------------
// Host-side orchestration
// ---------------------------------------------------------------------------
extern "C" void kernel_launch(void* const* d_in, const int* in_sizes, int n_in,
                              void* d_out, int out_size, void* d_ws, size_t ws_size,
                              hipStream_t stream) {
  (void)in_sizes; (void)n_in; (void)out_size; (void)ws_size;

  const int*   positives = (const int*)  d_in[0];
  const int*   mask      = (const int*)  d_in[1];
  /* d_in[2] = types (unused by reference) */
  const float* times     = (const float*)d_in[3];
  const float* item_emb  = (const float*)d_in[4];
  const float* pos_table = (const float*)d_in[5];
  const float* ln1_g = (const float*)d_in[6],  *ln1_b = (const float*)d_in[7];
  const float* Wq = (const float*)d_in[8],  *bq = (const float*)d_in[9];
  const float* Wk = (const float*)d_in[10], *bk = (const float*)d_in[11];
  const float* Wv = (const float*)d_in[12], *bv = (const float*)d_in[13];
  const float* Wo = (const float*)d_in[14], *bo = (const float*)d_in[15];
  const float* ln2_g = (const float*)d_in[16], *ln2_b = (const float*)d_in[17];
  const float* W1 = (const float*)d_in[18], *b1 = (const float*)d_in[19];
  const float* W2 = (const float*)d_in[20], *b2 = (const float*)d_in[21];
  const float* lnf_g = (const float*)d_in[22], *lnf_b = (const float*)d_in[23];

  // Workspace layout (bytes, all 256-aligned); ~160.4 MB total.
  char* ws = (char*)d_ws;
  float* x      = (float*)(ws + 0);                       //  33,554,432  residual stream fp32
  bf16*  ybuf   = (bf16*) (ws + 33554432ull);             //  16,777,216  LN out / attn out
  bf16*  qbuf   = (bf16*) (ws + 50331648ull);             //  16,777,216
  bf16*  kbuf   = (bf16*) (ws + 67108864ull);             //  16,777,216
  bf16*  vtbuf  = (bf16*) (ws + 83886080ull);             //  16,777,216  V^T (B, H, L)
  float* scores = (float*)(ws + 100663296ull);            //  67,108,864
  bf16*  probs  = qbuf;  // reuses q+k region (contiguous 33,554,432 B); q/k dead by then
  bf16*  wbf    = (bf16*) (ws + 167772160ull);            //     393,216  bf16 W^T weights

  const int HH = kH * kH;
  bf16* wq = wbf + 0 * kNL * HH;
  bf16* wk = wbf + 1 * kNL * HH;
  bf16* wv = wbf + 2 * kNL * HH;
  bf16* wo = wbf + 3 * kNL * HH;
  bf16* w1 = wbf + 4 * kNL * HH;
  bf16* w2 = wbf + 5 * kNL * HH;

  const int cvtN = kNL * HH;
  const int cvtB = (cvtN + 255) / 256;
  cvt_transpose_kernel<<<cvtB, 256, 0, stream>>>(Wq, wq, cvtN);
  cvt_transpose_kernel<<<cvtB, 256, 0, stream>>>(Wk, wk, cvtN);
  cvt_transpose_kernel<<<cvtB, 256, 0, stream>>>(Wv, wv, cvtN);
  cvt_transpose_kernel<<<cvtB, 256, 0, stream>>>(Wo, wo, cvtN);
  cvt_transpose_kernel<<<cvtB, 256, 0, stream>>>(W1, w1, cvtN);
  cvt_transpose_kernel<<<cvtB, 256, 0, stream>>>(W2, w2, cvtN);

  embed_rope_kernel<<<kBL, kH, 0, stream>>>(positives, times, item_emb, pos_table, x);

  dim3 gemmGrid(kBL / 64, kH / 32);   // (1024, 4)
  for (int i = 0; i < kNL; ++i) {
    // --- attention sub-block (norm_first) ---
    ln_kernel<bf16><<<kBL, kH, 0, stream>>>(x, ln1_g + i * kH, ln1_b + i * kH, ybuf);
    proj_gemm_kernel<<<gemmGrid, 128, 0, stream>>>(ybuf, wq + i * HH, bq + i * kH, qbuf, 0);
    proj_gemm_kernel<<<gemmGrid, 128, 0, stream>>>(ybuf, wk + i * HH, bk + i * kH, kbuf, 0);
    proj_gemm_kernel<<<gemmGrid, 128, 0, stream>>>(ybuf, wv + i * HH, bv + i * kH, vtbuf, 2);
    scores_kernel<<<dim3(kB, kL / 16, 4), 128, 0, stream>>>(qbuf, kbuf, scores);
    softmax_kernel<<<kBL, kL, 0, stream>>>(scores, mask, probs);
    attnv_kernel<<<dim3(kB, kL / 16), 128, 0, stream>>>(probs, vtbuf, ybuf);
    resid_gemm_kernel<<<gemmGrid, 128, 0, stream>>>(ybuf, wo + i * HH, bo + i * kH, x);
    // --- feed-forward sub-block ---
    ln_kernel<bf16><<<kBL, kH, 0, stream>>>(x, ln2_g + i * kH, ln2_b + i * kH, ybuf);
    proj_gemm_kernel<<<gemmGrid, 128, 0, stream>>>(ybuf, w1 + i * HH, b1 + i * kH, qbuf, 1);
    resid_gemm_kernel<<<gemmGrid, 128, 0, stream>>>(qbuf, w2 + i * HH, b2 + i * kH, x);
  }

  ln_kernel<float><<<kBL, kH, 0, stream>>>(x, lnf_g, lnf_b, (float*)d_out);
}
